// TinyRNN_53996328845814
// MI455X (gfx1250) — compile-verified, matrix operand order checked by reading, not run on hardware
//
#include <hip/hip_runtime.h>
#include <hip/hip_bf16.h>

#define B_ 128
#define T_ 512
#define I_ 512
#define H_ 1024
#define G_ (3 * H_)

// LDS row padding: +8 bf16 (16B) per row => row stride 1040B / 2064B.
// bank offset advances by 4 banks per row -> 16 rows span all 64 banks.
#define APAD1 (I_ + 8)     // 520  (gemm_fi A tile row stride, elements)
#define APAD2 (H_ + 8)     // 1032 (gru_step A tile row stride, elements)

typedef __attribute__((ext_vector_type(16))) __bf16 v16bf;
typedef __attribute__((ext_vector_type(8)))  __bf16 v8bf;
typedef __attribute__((ext_vector_type(8)))  float  v8f;

union ABFrag { v16bf v; v8bf h[2]; };

// ---------------------------------------------------------------------------
// Prep kernels
// ---------------------------------------------------------------------------
__global__ void f32_to_bf16_kernel(const float* __restrict__ src,
                                   __bf16* __restrict__ dst, long n) {
    long i = (long)blockIdx.x * blockDim.x + threadIdx.x;
    if (i < n) dst[i] = (__bf16)src[i];
}

// W: K x N (row-major f32)  ->  WT: N x K (row-major bf16)
__global__ void transpose_to_bf16_kernel(const float* __restrict__ W,
                                         __bf16* __restrict__ WT,
                                         int K, int N) {
    long i = (long)blockIdx.x * blockDim.x + threadIdx.x;
    if (i < (long)K * N) {
        int n = (int)(i / K);
        int k = (int)(i % K);
        WT[i] = (__bf16)W[(long)k * N + n];
    }
}

__global__ void init_h_kernel(float* __restrict__ hf, __bf16* __restrict__ hb) {
    int i = blockIdx.x * blockDim.x + threadIdx.x;
    if (i < B_ * H_) { hf[i] = 0.0f; hb[i] = (__bf16)0.0f; }
}

// ---------------------------------------------------------------------------
// Phase 1: fi = inputs @ W_in + bias_in  (bf16 in, f32 acc, bf16 out).
// Block = 8 waves sharing one 16-row M-tile; A tile staged in LDS once
// (coalesced), each wave owns one 16-col N-tile. K = 512.
// ---------------------------------------------------------------------------
__global__ void __launch_bounds__(256)
gemm_fi_kernel(const __bf16* __restrict__ X,      // (B*T) x I, bf16
               const __bf16* __restrict__ WinT,   // G x I (N-major), bf16
               const float*  __restrict__ bias,   // bias_in = bias[0..3H)
               __bf16* __restrict__ fi) {         // (B*T) x G, bf16
    __shared__ __attribute__((aligned(16))) __bf16 sA[16 * APAD1]; // 16.6 KB

    const int tid  = threadIdx.x;
    const int lane = tid & 31;
    const int warp = tid >> 5;
    const int sub  = lane >> 4;
    const int lm   = lane & 15;

    const int m_tile = blockIdx.x / (G_ / 16 / 8);   // 0..4095
    const int ngroup = blockIdx.x % (G_ / 16 / 8);   // 0..23
    const int m0 = m_tile * 16;
    const int n0 = (ngroup * 8 + warp) * 16;

    // Stage A tile: 16 x 512 bf16, fully coalesced (each thread 4 x 16B).
#pragma unroll
    for (int j = 0; j < 4; ++j) {
        int idx = tid * 32 + j * 8;          // 0..8191
        int row = idx >> 9;                  // /512
        int k   = idx & (I_ - 1);
        *(v8bf*)(&sA[row * APAD1 + k]) =
            *(const v8bf*)(X + (long)(m0 + row) * I_ + k);
    }
    __syncthreads();

    const __bf16* brow = WinT + (long)(n0 + lm) * I_;
    const __bf16* alds = &sA[lm * APAD1];

    v8f c = {};
#pragma unroll 4
    for (int k0 = 0; k0 < I_; k0 += 32) {
        ABFrag a, b;
        a.h[0] = *(const v8bf*)(alds + k0 + 8 * sub);
        a.h[1] = *(const v8bf*)(alds + k0 + 16 + 8 * sub);
        b.h[0] = *(const v8bf*)(brow + k0 + 8 * sub);
        b.h[1] = *(const v8bf*)(brow + k0 + 16 + 8 * sub);
        c = __builtin_amdgcn_wmma_f32_16x16x32_bf16(
                false, a.v, false, b.v, (short)0, c, false, false);
    }

    const float bia = bias[n0 + lm];
    __bf16* out = fi + (long)n0 + lm;
#pragma unroll
    for (int i = 0; i < 8; ++i) {
        int row = m0 + i + 8 * sub;                   // C layout: M = i + 8*sub
        out[(long)row * G_] = (__bf16)(c[i] + bia);
    }
}

// ---------------------------------------------------------------------------
// Phase 2: one launch per timestep. Block = 8 waves sharing one 16-batch
// tile; the h_{t-1} tile (16 x 1024 bf16) is staged in LDS once, each wave
// owns one 16-col h-tile and accumulates r/z/n with 3 WMMAs per K-step.
// ---------------------------------------------------------------------------
__global__ void __launch_bounds__(256)
gru_step_kernel(const __bf16* __restrict__ fi,      // (B*T) x G bf16
                const __bf16* __restrict__ WhT,     // G x H (N-major) bf16
                const float*  __restrict__ bias,    // full 6H bias
                const float*  __restrict__ hf_in,   // B x H f32 (h_{t-1})
                const __bf16* __restrict__ hb_in,   // B x H bf16 (h_{t-1})
                float*        __restrict__ hf_out,  // B x H f32 (h_t)
                __bf16*       __restrict__ hb_out,  // B x H bf16 (h_t)
                float*        __restrict__ out_seq, // B x T x H f32
                float*        __restrict__ out_last,// B x H f32
                int t) {
    __shared__ __attribute__((aligned(16))) __bf16 sA[16 * APAD2]; // 33 KB

    const int tid  = threadIdx.x;
    const int lane = tid & 31;
    const int warp = tid >> 5;
    const int sub  = lane >> 4;
    const int lm   = lane & 15;

    const int tile_m = blockIdx.x >> 3;              // 0..7  (batch tiles)
    const int ngroup = blockIdx.x & 7;               // 0..7
    const int m0 = tile_m * 16;
    const int n0 = (ngroup * 8 + warp) * 16;         // h-tile column base

    // Stage h_{t-1} tile: 16 x 1024 bf16, coalesced (each thread 8 x 16B).
#pragma unroll
    for (int j = 0; j < 8; ++j) {
        int idx = tid * 64 + j * 8;          // 0..16383
        int row = idx >> 10;                 // /1024
        int k   = idx & (H_ - 1);
        *(v8bf*)(&sA[row * APAD2 + k]) =
            *(const v8bf*)(hb_in + (long)(m0 + row) * H_ + k);
    }
    __syncthreads();

    const __bf16* alds = &sA[lm * APAD2];
    const __bf16* br = WhT + (long)(0 * H_ + n0 + lm) * H_;
    const __bf16* bz = WhT + (long)(1 * H_ + n0 + lm) * H_;
    const __bf16* bn = WhT + (long)(2 * H_ + n0 + lm) * H_;

    v8f cr = {}, cz = {}, cn = {};
#pragma unroll 2
    for (int k0 = 0; k0 < H_; k0 += 32) {
        ABFrag a, f0, f1, f2;
        a.h[0]  = *(const v8bf*)(alds + k0 + 8 * sub);
        a.h[1]  = *(const v8bf*)(alds + k0 + 16 + 8 * sub);
        f0.h[0] = *(const v8bf*)(br + k0 + 8 * sub);
        f0.h[1] = *(const v8bf*)(br + k0 + 16 + 8 * sub);
        f1.h[0] = *(const v8bf*)(bz + k0 + 8 * sub);
        f1.h[1] = *(const v8bf*)(bz + k0 + 16 + 8 * sub);
        f2.h[0] = *(const v8bf*)(bn + k0 + 8 * sub);
        f2.h[1] = *(const v8bf*)(bn + k0 + 16 + 8 * sub);
        cr = __builtin_amdgcn_wmma_f32_16x16x32_bf16(
                 false, a.v, false, f0.v, (short)0, cr, false, false);
        cz = __builtin_amdgcn_wmma_f32_16x16x32_bf16(
                 false, a.v, false, f1.v, (short)0, cz, false, false);
        cn = __builtin_amdgcn_wmma_f32_16x16x32_bf16(
                 false, a.v, false, f2.v, (short)0, cn, false, false);
    }

    const int   hcol = n0 + lm;
    const float bh_r = bias[3 * H_ + hcol];
    const float bh_z = bias[4 * H_ + hcol];
    const float bh_n = bias[5 * H_ + hcol];

#pragma unroll
    for (int i = 0; i < 8; ++i) {
        const int  brow   = m0 + i + 8 * sub;                 // batch index
        const long fibase = ((long)brow * T_ + t) * G_;
        const float fir = (float)fi[fibase + 0 * H_ + hcol];
        const float fiz = (float)fi[fibase + 1 * H_ + hcol];
        const float fin = (float)fi[fibase + 2 * H_ + hcol];

        const float fhr = cr[i] + bh_r;
        const float fhz = cz[i] + bh_z;
        const float fhn = cn[i] + bh_n;

        const float rt = 1.0f / (1.0f + __expf(-(fir + fhr)));
        const float zt = 1.0f / (1.0f + __expf(-(fiz + fhz)));
        const float nt = tanhf(fin + rt * fhn);

        const float hp = hf_in[(long)brow * H_ + hcol];
        const float ht = (1.0f - zt) * nt + zt * hp;

        hf_out[(long)brow * H_ + hcol] = ht;
        hb_out[(long)brow * H_ + hcol] = (__bf16)ht;
        out_seq[((long)brow * T_ + t) * H_ + hcol] = ht;
        if (t == T_ - 1) out_last[(long)brow * H_ + hcol] = ht;
    }
}

// ---------------------------------------------------------------------------
extern "C" void kernel_launch(void* const* d_in, const int* in_sizes, int n_in,
                              void* d_out, int out_size, void* d_ws, size_t ws_size,
                              hipStream_t stream) {
    const float* inputs = (const float*)d_in[0];   // B x T x I
    const float* Win    = (const float*)d_in[1];   // I x 3H
    const float* Wh     = (const float*)d_in[2];   // H x 3H
    const float* bias   = (const float*)d_in[3];   // 6H

    float* out      = (float*)d_out;
    float* out_last = out + (long)B_ * T_ * H_;

    // Workspace carve-up (all chunk sizes are multiples of 16B).
    char*   ws   = (char*)d_ws;
    size_t  off  = 0;
    __bf16* WinT = (__bf16*)(ws + off); off += (size_t)G_ * I_ * 2;       //  3 MB
    __bf16* WhT  = (__bf16*)(ws + off); off += (size_t)G_ * H_ * 2;       //  6 MB
    __bf16* Xb   = (__bf16*)(ws + off); off += (size_t)B_ * T_ * I_ * 2;  // 67 MB
    __bf16* fi   = (__bf16*)(ws + off); off += (size_t)B_ * T_ * G_ * 2;  // 403 MB
    float*  hf0  = (float*)(ws + off);  off += (size_t)B_ * H_ * 4;
    float*  hf1  = (float*)(ws + off);  off += (size_t)B_ * H_ * 4;
    __bf16* hb0  = (__bf16*)(ws + off); off += (size_t)B_ * H_ * 2;
    __bf16* hb1  = (__bf16*)(ws + off); off += (size_t)B_ * H_ * 2;

    // Prep: bf16 conversions + weight transposes.
    {
        long n = (long)G_ * I_;
        transpose_to_bf16_kernel<<<(int)((n + 255) / 256), 256, 0, stream>>>(Win, WinT, I_, G_);
    }
    {
        long n = (long)G_ * H_;
        transpose_to_bf16_kernel<<<(int)((n + 255) / 256), 256, 0, stream>>>(Wh, WhT, H_, G_);
    }
    {
        long n = (long)B_ * T_ * I_;
        f32_to_bf16_kernel<<<(int)((n + 255) / 256), 256, 0, stream>>>(inputs, Xb, n);
    }

    // Phase 1: 4096 M-tiles x 24 N-groups (8 waves each).
    gemm_fi_kernel<<<4096 * (G_ / 16 / 8), 256, 0, stream>>>(Xb, WinT, bias, fi);

    // h0 = 0
    init_h_kernel<<<(B_ * H_ + 255) / 256, 256, 0, stream>>>(hf0, hb0);

    // Phase 2: sequential scan, one launch per step (graph-captured).
    for (int t = 0; t < T_; ++t) {
        const float*  hin   = (t & 1) ? hf1 : hf0;
        const __bf16* hbin  = (t & 1) ? hb1 : hb0;
        float*        hout  = (t & 1) ? hf0 : hf1;
        __bf16*       hbout = (t & 1) ? hb0 : hb1;
        gru_step_kernel<<<64, 256, 0, stream>>>(fi, WhT, bias, hin, hbin,
                                                hout, hbout, out, out_last, t);
    }
}
